// DecoderLayer_11373073400217
// MI455X (gfx1250) — matrix operand order repaired and checked by reading, not compiled, not run on hardware
//
#include <hip/hip_runtime.h>
#include <cstddef>
#include <cstdint>

// ---------------------------------------------------------------------------
// Types for CDNA5 WMMA (wave32): 16x16x32 bf16 -> f32, plus TDM descriptors
// ---------------------------------------------------------------------------
typedef __attribute__((ext_vector_type(16))) __bf16 v16bf;
typedef __attribute__((ext_vector_type(8)))  __bf16 bf16x8;
typedef __attribute__((ext_vector_type(8)))  float  f32x8;
typedef __attribute__((ext_vector_type(4)))  unsigned int v4u;
typedef __attribute__((ext_vector_type(8)))  int v8i;
typedef __attribute__((ext_vector_type(4)))  int v4i;

typedef __attribute__((address_space(3))) __bf16 lds_bf16_t;

__device__ __forceinline__ f32x8 wmma_bf16(v16bf a, v16bf b, f32x8 c) {
  return __builtin_amdgcn_wmma_f32_16x16x32_bf16(false, a, false, b, (short)0, c,
                                                 false, false);
}

// A-fragment: 16x32 bf16 tile, row m = lane&15.
// Lanes 0-15 hold K = 0..15; lanes 16-31 hold K = 16..31.
__device__ __forceinline__ v16bf load_a_frag(const __bf16* tile, int ld, int k0) {
  int lane = threadIdx.x & 31;
  const __bf16* p = tile + (size_t)(lane & 15) * ld + k0 + ((lane >> 4) << 4);
  bf16x8 lo = *(const bf16x8*)p;
  bf16x8 hi = *(const bf16x8*)(p + 8);
  v16bf f;
#pragma unroll
  for (int i = 0; i < 8; ++i) { f[i] = lo[i]; f[8 + i] = hi[i]; }
  return f;
}

// B-fragment from an [N][K] row-major source: col n = lane&15;
// lanes 0-15: K=k0..k0+15, lanes 16-31: K=k0+16..k0+31 (one 32B run).
__device__ __forceinline__ v16bf load_b_frag_nk(const __bf16* tile, int ld, int k0) {
  int lane = threadIdx.x & 31;
  const __bf16* p = tile + (size_t)(lane & 15) * ld + k0 + ((lane >> 4) << 4);
  return *(const v16bf*)p;
}

// B-fragment from a [K][N] row-major source (attention V).
__device__ __forceinline__ v16bf load_b_frag_kn(const __bf16* tile, int ld) {
  int lane = threadIdx.x & 31;
  int nn = lane & 15;
  int kb = (lane >> 4) << 4;
  v16bf f;
#pragma unroll
  for (int v = 0; v < 8; ++v) {
    f[2 * v]     = tile[(size_t)(kb + 2 * v) * ld + nn];
    f[2 * v + 1] = tile[(size_t)(kb + 2 * v + 1) * ld + nn];
  }
  return f;
}

// ---------------------------------------------------------------------------
// Tensor Data Mover: 2-D tile load (rows x cols of bf16) global -> LDS.
// D# per CDNA5 ISA ch.8: group0 = {count/flags, lds_addr, global_addr, type},
// group1 = {mask/data_size/pad, dims/tile dims/strides}. Groups 2/3 zero
// (2-D tensor: tile_dim2 = 0 -> unused). 6-arg clang-23 builtin form.
// ---------------------------------------------------------------------------
__device__ __forceinline__ void tdm_load_tile_2d(const __bf16* g, uint64_t ldElems,
                                                 __bf16* lds, uint32_t tileCols,
                                                 uint32_t tileRows) {
  uint64_t ga = (uint64_t)(uintptr_t)g;
  uint32_t loff = (uint32_t)(uintptr_t)(lds_bf16_t*)lds;  // LDS byte offset
  v4u g0;
  g0[0] = 1u;                                   // count=1, user descriptor
  g0[1] = loff;                                 // lds_addr
  g0[2] = (uint32_t)ga;                         // global_addr[31:0]
  g0[3] = (uint32_t)((ga >> 32) & 0x01FFFFFFu) | (2u << 30);  // addr hi | type=2
  uint32_t dim0 = (uint32_t)ldElems;            // tensor dim0 (elements)
  uint32_t dim1 = 1u << 20;                     // tensor dim1 (ample; in-bounds use)
  uint64_t s0 = ldElems;                        // dim0 stride (elements)
  v8i g1;
  g1[0] = 0x10000;                              // data_size=1 -> 2 bytes/elem
  g1[1] = (int)((dim0 & 0xFFFFu) << 16);        // tensor_dim0[15:0] @ bits 63:48
  g1[2] = (int)((dim0 >> 16) | ((dim1 & 0xFFFFu) << 16));
  g1[3] = (int)((dim1 >> 16) | (tileCols << 16));   // tile_dim0 @ 127:112
  g1[4] = (int)(tileRows & 0xFFFFu);                // tile_dim1; tile_dim2=0
  g1[5] = (int)(s0 & 0xFFFFFFFFu);                  // dim0_stride[31:0]
  g1[6] = (int)((s0 >> 32) & 0xFFFFu);              // dim0_stride[47:32]
  g1[7] = 0;
  v4i z4 = {0, 0, 0, 0};
  v8i z8 = {0, 0, 0, 0, 0, 0, 0, 0};
  __builtin_amdgcn_tensor_load_to_lds(g0, g1, z4, z4, z8, 0);
}

// ---------------------------------------------------------------------------
// Elementwise converts
// ---------------------------------------------------------------------------
__global__ void f2b_kernel(const float* __restrict__ s, __bf16* __restrict__ d,
                           size_t n) {
  size_t i = (size_t)blockIdx.x * blockDim.x + threadIdx.x;
  size_t st = (size_t)gridDim.x * blockDim.x;
  for (; i < n; i += st) d[i] = (__bf16)s[i];
}

__global__ void add_f2b_kernel(const float* __restrict__ a,
                               const float* __restrict__ b,
                               __bf16* __restrict__ d, size_t n) {
  size_t i = (size_t)blockIdx.x * blockDim.x + threadIdx.x;
  size_t st = (size_t)gridDim.x * blockDim.x;
  for (; i < n; i += st) d[i] = (__bf16)(a[i] + b[i]);
}

// ---------------------------------------------------------------------------
// TDM-fed LDS-tiled GEMM: out = epi(A[MxK]bf16 @ Wt[NxK]^T + bias + residual).
// Block tile 128x128, K chunk 64, double-buffered LDS (64 KB). Wave 0 issues
// two tensor_load_to_lds per chunk; 8 waves each compute a 32x64 subtile from
// LDS. All fragment loads for one K-step are hoisted ahead of the 8 WMMAs so
// the ds_load_b128s issue as one clause and drain with partial dscnt waits.
// M%128==0, N%128==0, K%64==0 (true for all calls here).
// ---------------------------------------------------------------------------
__global__ __launch_bounds__(256) void gemm_tdm_kernel(
    const __bf16* __restrict__ A, const __bf16* __restrict__ Wt,
    const float* __restrict__ bias, const float* __restrict__ residual,
    float* __restrict__ outF, __bf16* __restrict__ outB,
    int M, int N, int K, int relu) {
  __shared__ __bf16 sA[2][128 * 64];
  __shared__ __bf16 sB[2][128 * 64];
  const int KC = 64;
  int nbN = N >> 7;
  int m0 = (blockIdx.x / nbN) << 7;
  int n0 = (blockIdx.x % nbN) << 7;
  int wv = threadIdx.x >> 5;
  int mw = (wv >> 1) << 5;   // 0,32,64,96
  int nw = (wv & 1) << 6;    // 0,64
  bool issuer = (wv == 0);
  int nk = K / KC;

  if (issuer) {
    tdm_load_tile_2d(A + (size_t)m0 * K, (uint64_t)K, &sA[0][0], KC, 128);
    tdm_load_tile_2d(Wt + (size_t)n0 * K, (uint64_t)K, &sB[0][0], KC, 128);
    __builtin_amdgcn_s_wait_tensorcnt(0);
  }
  __syncthreads();

  f32x8 acc[2][4] = {};
  for (int i = 0; i < nk; ++i) {
    int cur = i & 1;
    if (issuer && (i + 1) < nk) {  // prefetch next chunk while computing
      tdm_load_tile_2d(A + (size_t)m0 * K + (size_t)(i + 1) * KC, (uint64_t)K,
                       &sA[cur ^ 1][0], KC, 128);
      tdm_load_tile_2d(Wt + (size_t)n0 * K + (size_t)(i + 1) * KC, (uint64_t)K,
                       &sB[cur ^ 1][0], KC, 128);
    }
    const __bf16* At = &sA[cur][0] + (size_t)mw * KC;
    const __bf16* Bt = &sB[cur][0] + (size_t)nw * KC;
#pragma unroll
    for (int ks = 0; ks < 2; ++ks) {
      int k0 = ks * 32;
      // hoist all 6 fragment loads ahead of the 8 WMMAs (distinct live regs)
      v16bf a0 = load_a_frag(At, KC, k0);
      v16bf a1 = load_a_frag(At + 16 * KC, KC, k0);
      v16bf b0 = load_b_frag_nk(Bt + (size_t)0 * 16 * KC, KC, k0);
      v16bf b1 = load_b_frag_nk(Bt + (size_t)1 * 16 * KC, KC, k0);
      v16bf b2 = load_b_frag_nk(Bt + (size_t)2 * 16 * KC, KC, k0);
      v16bf b3 = load_b_frag_nk(Bt + (size_t)3 * 16 * KC, KC, k0);
      acc[0][0] = wmma_bf16(a0, b0, acc[0][0]);
      acc[1][0] = wmma_bf16(a1, b0, acc[1][0]);
      acc[0][1] = wmma_bf16(a0, b1, acc[0][1]);
      acc[1][1] = wmma_bf16(a1, b1, acc[1][1]);
      acc[0][2] = wmma_bf16(a0, b2, acc[0][2]);
      acc[1][2] = wmma_bf16(a1, b2, acc[1][2]);
      acc[0][3] = wmma_bf16(a0, b3, acc[0][3]);
      acc[1][3] = wmma_bf16(a1, b3, acc[1][3]);
    }
    if (issuer) __builtin_amdgcn_s_wait_tensorcnt(0);
    __syncthreads();
  }

  int lane = threadIdx.x & 31, g = lane >> 4, nn = lane & 15;
#pragma unroll
  for (int mi = 0; mi < 2; ++mi) {
#pragma unroll
    for (int t = 0; t < 4; ++t) {
      int col = n0 + nw + t * 16 + nn;
      float bv = bias ? bias[col] : 0.f;
#pragma unroll
      for (int r = 0; r < 8; ++r) {
        int row = m0 + mw + mi * 16 + r + 8 * g;
        float v = acc[mi][t][r] + bv;
        size_t idx = (size_t)row * N + col;
        if (residual) v += residual[idx];
        if (relu) v = fmaxf(v, 0.f);
        if (outF) outF[idx] = v;
        if (outB) outB[idx] = (__bf16)v;
      }
    }
  }
}

// ---------------------------------------------------------------------------
// Flash attention: one wave per (batch, head, 16-query tile). head_dim=64,
// row stride 1024. Streaming softmax over keys in chunks of 32; P tile
// round-trips through LDS (C-layout -> A-layout).
// ---------------------------------------------------------------------------
__global__ __launch_bounds__(256) void attn_kernel(
    const __bf16* __restrict__ Q, const __bf16* __restrict__ Km,
    const __bf16* __restrict__ Vm, __bf16* __restrict__ O,
    int nkeys, int kstride, float scale, int totalWaves) {
  __shared__ __bf16 ptile[8][16 * 32];
  int wv = threadIdx.x >> 5;
  int wave = blockIdx.x * (blockDim.x >> 5) + wv;
  if (wave >= totalWaves) return;

  const int Wd = 1024, HD = 64, NQc = 256, NQt = 16, NHc = 16;
  int qt = wave % NQt;
  int h  = (wave / NQt) % NHc;
  int b  = wave / (NQt * NHc);

  const __bf16* Qb = Q + ((size_t)b * NQc + qt * 16) * Wd + h * HD;
  const __bf16* Kb = Km + (size_t)b * kstride * Wd + h * HD;
  const __bf16* Vb = Vm + (size_t)b * kstride * Wd + h * HD;

  v16bf qa0 = load_a_frag(Qb, Wd, 0);
  v16bf qa1 = load_a_frag(Qb, Wd, 32);

  f32x8 acc[4] = {};
  float rmax[8], rsum[8];
#pragma unroll
  for (int r = 0; r < 8; ++r) { rmax[r] = -3.0e38f; rsum[r] = 0.f; }

  int lane = threadIdx.x & 31, g = lane >> 4, nn = lane & 15;
  __bf16* tile = &ptile[wv][0];

  for (int j = 0; j < nkeys; j += 32) {
    f32x8 s0 = {}, s1 = {};
    s0 = wmma_bf16(qa0, load_b_frag_nk(Kb + (size_t)j * Wd, Wd, 0), s0);
    s0 = wmma_bf16(qa1, load_b_frag_nk(Kb + (size_t)j * Wd, Wd, 32), s0);
    s1 = wmma_bf16(qa0, load_b_frag_nk(Kb + (size_t)(j + 16) * Wd, Wd, 0), s1);
    s1 = wmma_bf16(qa1, load_b_frag_nk(Kb + (size_t)(j + 16) * Wd, Wd, 32), s1);

#pragma unroll
    for (int r = 0; r < 8; ++r) {
      float a = s0[r] * scale, c = s1[r] * scale;
      float mloc = fmaxf(a, c);
      mloc = fmaxf(mloc, __shfl_xor(mloc, 1, 32));
      mloc = fmaxf(mloc, __shfl_xor(mloc, 2, 32));
      mloc = fmaxf(mloc, __shfl_xor(mloc, 4, 32));
      mloc = fmaxf(mloc, __shfl_xor(mloc, 8, 32));
      float mnew = fmaxf(rmax[r], mloc);
      float e = __expf(rmax[r] - mnew);
      rmax[r] = mnew;
      float p0 = __expf(a - mnew), p1 = __expf(c - mnew);
      float ps = p0 + p1;
      ps += __shfl_xor(ps, 1, 32);
      ps += __shfl_xor(ps, 2, 32);
      ps += __shfl_xor(ps, 4, 32);
      ps += __shfl_xor(ps, 8, 32);
      rsum[r] = rsum[r] * e + ps;
#pragma unroll
      for (int t = 0; t < 4; ++t) acc[t][r] *= e;
      tile[(r + 8 * g) * 32 + nn]      = (__bf16)p0;
      tile[(r + 8 * g) * 32 + nn + 16] = (__bf16)p1;
    }
    asm volatile("s_wait_dscnt 0" ::: "memory");  // LDS RAW across lanes
    v16bf pf = load_a_frag(tile, 32, 0);
    // hoist the 4 V fragments so their loads overlap the P-fragment wait
    v16bf v0 = load_b_frag_kn(Vb + (size_t)j * Wd + 0, Wd);
    v16bf v1 = load_b_frag_kn(Vb + (size_t)j * Wd + 16, Wd);
    v16bf v2 = load_b_frag_kn(Vb + (size_t)j * Wd + 32, Wd);
    v16bf v3 = load_b_frag_kn(Vb + (size_t)j * Wd + 48, Wd);
    acc[0] = wmma_bf16(pf, v0, acc[0]);
    acc[1] = wmma_bf16(pf, v1, acc[1]);
    acc[2] = wmma_bf16(pf, v2, acc[2]);
    acc[3] = wmma_bf16(pf, v3, acc[3]);
  }

  __bf16* Ob = O + ((size_t)b * NQc + qt * 16) * Wd + h * HD;
#pragma unroll
  for (int t = 0; t < 4; ++t)
#pragma unroll
    for (int r = 0; r < 8; ++r)
      Ob[(size_t)(r + 8 * g) * Wd + t * 16 + nn] = (__bf16)(acc[t][r] / rsum[r]);
}

// ---------------------------------------------------------------------------
// LayerNorm over rows of width 1024 (one block of 256 threads per row).
// ---------------------------------------------------------------------------
__device__ __forceinline__ float block_sum(float v, float* sm) {
#pragma unroll
  for (int m = 16; m > 0; m >>= 1) v += __shfl_xor(v, m, 32);
  int lane = threadIdx.x & 31, wv = threadIdx.x >> 5;
  if (lane == 0) sm[wv] = v;
  __syncthreads();
  float r = sm[0];
#pragma unroll
  for (int i = 1; i < 8; ++i) r += sm[i];
  __syncthreads();
  return r;
}

__global__ __launch_bounds__(256) void ln_kernel(
    const float* __restrict__ x, const float* __restrict__ gam,
    const float* __restrict__ bet, float* __restrict__ outF,
    __bf16* __restrict__ outB) {
  const int Wd = 1024;
  __shared__ float sm[8];
  size_t row = blockIdx.x;
  const float* xr = x + row * Wd;
  float v[4], s = 0.f;
#pragma unroll
  for (int i = 0; i < 4; ++i) { v[i] = xr[threadIdx.x + 256 * i]; s += v[i]; }
  float mean = block_sum(s, sm) * (1.f / Wd);
  float var = 0.f;
#pragma unroll
  for (int i = 0; i < 4; ++i) { float d = v[i] - mean; var += d * d; }
  var = block_sum(var, sm) * (1.f / Wd);
  float inv = rsqrtf(var + 1e-5f);
#pragma unroll
  for (int i = 0; i < 4; ++i) {
    int c = threadIdx.x + 256 * i;
    float y = (v[i] - mean) * inv * gam[c] + bet[c];
    if (outF) outF[row * Wd + c] = y;
    if (outB) outB[row * Wd + c] = (__bf16)y;
  }
}

// ---------------------------------------------------------------------------
// Host orchestration
// ---------------------------------------------------------------------------
extern "C" void kernel_launch(void* const* d_in, const int* in_sizes, int n_in,
                              void* d_out, int out_size, void* d_ws,
                              size_t ws_size, hipStream_t stream) {
  (void)in_sizes; (void)n_in; (void)out_size; (void)ws_size;
  const int Wd = 1024, MLPd = 4096, NQ = 256, S = 4096, Bc = 16;
  const size_t M1 = (size_t)Bc * NQ;   // 4096
  const size_t M2 = (size_t)Bc * S;    // 65536

  const float* query = (const float*)d_in[0];
  const float* enc   = (const float*)d_in[1];
  const float* pos   = (const float*)d_in[2];
  const float* wsrc[10] = {
      (const float*)d_in[3], (const float*)d_in[4], (const float*)d_in[5],
      (const float*)d_in[6], (const float*)d_in[7], (const float*)d_in[8],
      (const float*)d_in[9], (const float*)d_in[10], (const float*)d_in[11],
      (const float*)d_in[13]};
  const float* b1 = (const float*)d_in[12];
  const float* b2 = (const float*)d_in[14];
  const float* ln1g = (const float*)d_in[15];
  const float* ln1b = (const float*)d_in[16];
  const float* ln2g = (const float*)d_in[17];
  const float* ln2b = (const float*)d_in[18];
  const float* ln3g = (const float*)d_in[19];
  const float* ln3b = (const float*)d_in[20];

  char* ws = (char*)d_ws;
  size_t off = 0;
  auto take = [&](size_t bytes) -> void* {
    void* p = ws + off;
    off += (bytes + 255) & ~(size_t)255;
    return p;
  };
  __bf16* enc_bf = (__bf16*)take(M2 * Wd * 2);  // 128 MiB, recycled as pool
  __bf16* caK    = (__bf16*)take(M2 * Wd * 2);
  __bf16* caV    = (__bf16*)take(M2 * Wd * 2);
  float*  x1f    = (float*)take(M1 * Wd * 4);
  float*  x2f    = (float*)take(M1 * Wd * 4);
  float*  tmpf   = (float*)take(M1 * Wd * 4);
  __bf16* wb     = (__bf16*)take((8ull * 1024 * 1024 + 2ull * 4096 * 1024) * 2);

  __bf16* wdst[10];
  size_t wcnt[10];
  {
    size_t o = 0;
    for (int i = 0; i < 8; ++i) { wdst[i] = wb + o; wcnt[i] = 1048576; o += 1048576; }
    wdst[8] = wb + o; wcnt[8] = 4194304; o += 4194304;   // ffn_w1 [4096][1024]
    wdst[9] = wb + o; wcnt[9] = 4194304;                 // ffn_w2 [1024][4096]
  }

  // pool slots (8 MiB each) overlaid on enc_bf after cross K/V are projected
  char* pool = (char*)enc_bf;
  auto slot = [&](int i) { return (__bf16*)(pool + (size_t)i * 8388608); };
  __bf16 *qk_bf = slot(0), *q_bf = slot(1), *saq = slot(2), *sak = slot(3),
         *sav = slot(4), *attnout = slot(5), *ca_qin = slot(6), *caq = slot(7),
         *ca_attnout = slot(8), *x2bf = slot(9), *ffn_h = slot(10);  // 10..13

  auto gemm = [&](const __bf16* A, const __bf16* Wt, const float* bias,
                  const float* res, float* oF, __bf16* oB, int M, int N, int K,
                  int relu) {
    int blocks = (M >> 7) * (N >> 7);
    gemm_tdm_kernel<<<blocks, 256, 0, stream>>>(A, Wt, bias, res, oF, oB, M, N,
                                                K, relu);
  };

  const float scale = 0.03125f;  // W^-0.5

  // Phase 0: converts + cross-attn K/V projections (then enc_bf is recycled)
  f2b_kernel<<<4096, 256, 0, stream>>>(enc, enc_bf, M2 * Wd);
  for (int i = 0; i < 10; ++i)
    f2b_kernel<<<512, 256, 0, stream>>>(wsrc[i], wdst[i], wcnt[i]);
  gemm(enc_bf, wdst[5], nullptr, nullptr, nullptr, caK, (int)M2, Wd, Wd, 0);
  gemm(enc_bf, wdst[6], nullptr, nullptr, nullptr, caV, (int)M2, Wd, Wd, 0);

  // Self-attention: qk = query+pos for Q,K; v from query
  add_f2b_kernel<<<1024, 256, 0, stream>>>(query, pos, qk_bf, M1 * Wd);
  f2b_kernel<<<1024, 256, 0, stream>>>(query, q_bf, M1 * Wd);
  gemm(qk_bf, wdst[0], nullptr, nullptr, nullptr, saq, (int)M1, Wd, Wd, 0);
  gemm(qk_bf, wdst[1], nullptr, nullptr, nullptr, sak, (int)M1, Wd, Wd, 0);
  gemm(q_bf,  wdst[2], nullptr, nullptr, nullptr, sav, (int)M1, Wd, Wd, 0);
  attn_kernel<<<512, 256, 0, stream>>>(saq, sak, sav, attnout, NQ, NQ, scale, 4096);
  gemm(attnout, wdst[3], nullptr, query, tmpf, nullptr, (int)M1, Wd, Wd, 0);
  ln_kernel<<<(int)M1, 256, 0, stream>>>(tmpf, ln1g, ln1b, x1f, nullptr);

  // Cross-attention
  add_f2b_kernel<<<1024, 256, 0, stream>>>(x1f, pos, ca_qin, M1 * Wd);
  gemm(ca_qin, wdst[4], nullptr, nullptr, nullptr, caq, (int)M1, Wd, Wd, 0);
  attn_kernel<<<512, 256, 0, stream>>>(caq, caK, caV, ca_attnout, S, S, scale, 4096);
  gemm(ca_attnout, wdst[7], nullptr, x1f, tmpf, nullptr, (int)M1, Wd, Wd, 0);
  ln_kernel<<<(int)M1, 256, 0, stream>>>(tmpf, ln2g, ln2b, x2f, x2bf);

  // FFN
  gemm(x2bf, wdst[8], b1, nullptr, nullptr, ffn_h, (int)M1, MLPd, Wd, 1);
  gemm(ffn_h, wdst[9], b2, x2f, tmpf, nullptr, (int)M1, Wd, MLPd, 0);
  ln_kernel<<<(int)M1, 256, 0, stream>>>(tmpf, ln3g, ln3b, (float*)d_out, nullptr);
}
